// MambaBlock_57346403336598
// MI455X (gfx1250) — compile-verified, matrix operand order checked by reading, not run on hardware
//
#include <hip/hip_runtime.h>
#include <hip/hip_bf16.h>
#include <stdint.h>

// Problem constants (from reference)
#define BDIM   4
#define LDIM   1024
#define DMODEL 1024
#define EDIM   2048
#define NSTATE 16
#define KCONV  4
#define RRANK  64
#define BL     (BDIM * LDIM)
#define DBCW   (RRANK + 2 * NSTATE)   // 96

typedef __attribute__((ext_vector_type(16))) __bf16 v16bf;
typedef __attribute__((ext_vector_type(8)))  float  v8f;

static __device__ __forceinline__ unsigned short f2bf(float f) {
    union { float f; uint32_t u; } a; a.f = f;
    uint32_t u = a.u;
    uint32_t r = u + 0x7FFFu + ((u >> 16) & 1u);   // round-to-nearest-even
    return (unsigned short)(r >> 16);
}

static __device__ __forceinline__ float silu(float x) {
    return x / (1.0f + __expf(-x));
}

// async copy 16B global -> LDS (no VGPR data path, tracked by ASYNCcnt)
static __device__ __forceinline__ void async_cp16(uint32_t lds_off,
                                                  const void* gaddr) {
    asm volatile("global_load_async_to_lds_b128 %0, %1, off"
                 :: "v"(lds_off), "v"((unsigned long long)(uintptr_t)gaddr)
                 : "memory");
}
static __device__ __forceinline__ void wait_async0() {
    asm volatile("s_wait_asynccnt 0x0" ::: "memory");
}

// ---------------------------------------------------------------------------
// fp32 -> bf16 conversion, 4 elements per thread (n must be multiple of 4)
// ---------------------------------------------------------------------------
__global__ void cvt_f32_to_bf16x4(const float* __restrict__ src,
                                  unsigned short* __restrict__ dst, int n4) {
    int i = blockIdx.x * 256 + threadIdx.x;
    if (i >= n4) return;
    const float4 v = ((const float4*)src)[i];
    ushort4 o;
    o.x = f2bf(v.x); o.y = f2bf(v.y); o.z = f2bf(v.z); o.w = f2bf(v.w);
    ((ushort4*)dst)[i] = o;
}

// ---------------------------------------------------------------------------
// bf16 WMMA GEMM: C[M,N] = A[M,K] * B[N,K]^T   (both A,B row-major, K inner)
// Block: 256 threads (8 waves), tile 128(M) x 64(N), K-chunk 64.
// Double-buffered LDS fed by ASYNC global->LDS copies (no staging VGPRs,
// no scratch spills possible):
//   - issue global_load_async_to_lds_b128 of tile k+1 into LDS[buf^1]
//   - WMMA on LDS[buf] while the async copies are in flight
//   - s_wait_asynccnt 0 ; one barrier per K-chunk ; flip
// OOB B-rows (N edge) are zero-filled once in the prologue (skipped async
// lanes never overwrite them).
// Requires M % 128 == 0, K % 64 == 0. N handled with guards.
// ---------------------------------------------------------------------------
#define GBM 128
#define GBN 64
#define GBK 64
#define GPA 72   // LDS pitch in bf16 elems (144B rows; 16B aligned, conflict-free)

__launch_bounds__(256)
__global__ void gemm_bf16_wmma(const unsigned short* __restrict__ A,
                               const unsigned short* __restrict__ Bm,
                               float* __restrict__ C,
                               int M, int N, int K) {
    __shared__ __align__(16) unsigned short As[2][GBM * GPA];   // 2 x 18432 B
    __shared__ __align__(16) unsigned short Bs[2][GBN * GPA];   // 2 x  9216 B

    const int tid  = threadIdx.x;
    const int lane = tid & 31;
    const int w    = tid >> 5;
    const int wm   = w & 3;        // 0..3 -> M sub-tile
    const int wn   = w >> 2;       // 0..1 -> N sub-tile
    const int m0   = blockIdx.y * GBM;
    const int n0   = blockIdx.x * GBN;

    const int r16 = lane & 15;
    const int klo = (lane >> 4) << 3;   // 0 or 8: K-half per ISA 16-bit layout

    // LDS byte offsets of the buffers (aperture is in the high 32 bits)
    const uint32_t as_base = (uint32_t)(uintptr_t)(&As[0][0]);
    const uint32_t bs_base = (uint32_t)(uintptr_t)(&Bs[0][0]);

    v8f acc[2][2];
    #pragma unroll
    for (int i = 0; i < 2; i++)
        #pragma unroll
        for (int j = 0; j < 2; j++)
            acc[i][j] = {};

    // A tile 128x64 bf16 = 1024 x 16B chunks -> 4 per thread
    // B tile  64x64 bf16 =  512 x 16B chunks -> 2 per thread
    auto issue_async = [&](int buf, int k0) {
        #pragma unroll
        for (int i = 0; i < 4; i++) {
            int c = tid + i * 256;
            int row = c >> 3, cc = (c & 7) << 3;
            async_cp16(as_base + (uint32_t)(buf * (GBM * GPA) + row * GPA + cc) * 2u,
                       A + (size_t)(m0 + row) * K + k0 + cc);
        }
        #pragma unroll
        for (int i = 0; i < 2; i++) {
            int c = tid + i * 256;
            int row = c >> 3, cc = (c & 7) << 3;
            if (n0 + row < N)
                async_cp16(bs_base + (uint32_t)(buf * (GBN * GPA) + row * GPA + cc) * 2u,
                           Bm + (size_t)(n0 + row) * K + k0 + cc);
        }
    };
    auto compute = [&](int buf) {
        union Frag { v16bf v; uint4 u[2]; };
        #pragma unroll
        for (int kk = 0; kk < GBK; kk += 32) {
            Frag af[2], bfr[2];
            #pragma unroll
            for (int i = 0; i < 2; i++) {
                int row = wm * 32 + i * 16 + r16;
                af[i].u[0] = *(const uint4*)(&As[buf][row * GPA + kk + klo]);
                af[i].u[1] = *(const uint4*)(&As[buf][row * GPA + kk + klo + 16]);
            }
            #pragma unroll
            for (int j = 0; j < 2; j++) {
                int row = wn * 32 + j * 16 + r16;
                bfr[j].u[0] = *(const uint4*)(&Bs[buf][row * GPA + kk + klo]);
                bfr[j].u[1] = *(const uint4*)(&Bs[buf][row * GPA + kk + klo + 16]);
            }
            #pragma unroll
            for (int i = 0; i < 2; i++)
                #pragma unroll
                for (int j = 0; j < 2; j++)
                    acc[i][j] = __builtin_amdgcn_wmma_f32_16x16x32_bf16(
                        false, af[i].v, false, bfr[j].v,
                        (short)0, acc[i][j], false, false);
        }
    };

    // prologue: zero OOB B rows in BOTH buffers (async copies skip them)
    #pragma unroll
    for (int i = 0; i < 2; i++) {
        int c = tid + i * 256;
        int row = c >> 3, cc = (c & 7) << 3;
        if (n0 + row >= N) {
            uint4 z = make_uint4(0u, 0u, 0u, 0u);
            *(uint4*)(&Bs[0][row * GPA + cc]) = z;
            *(uint4*)(&Bs[1][row * GPA + cc]) = z;
        }
    }
    issue_async(0, 0);
    wait_async0();
    __syncthreads();

    int buf = 0;
    for (int k0 = GBK; k0 < K; k0 += GBK) {
        issue_async(buf ^ 1, k0);   // async copies into the other buffer
        compute(buf);               // WMMA while copies are in flight
        wait_async0();              // this wave's copies into buf^1 done
        __syncthreads();            // all waves done: publish buf^1, retire buf
        buf ^= 1;
    }
    compute(buf);

    // Store: VGPR r -> row r (lanes 0-15) / r+8 (lanes 16-31); col = lane&15
    const int rowadd = (lane >> 4) << 3;
    #pragma unroll
    for (int i = 0; i < 2; i++) {
        int rb = m0 + wm * 32 + i * 16 + rowadd;
        #pragma unroll
        for (int j = 0; j < 2; j++) {
            int col = n0 + wn * 32 + j * 16 + (lane & 15);
            if (col < N) {
                #pragma unroll
                for (int r = 0; r < 8; r++) {
                    int row = rb + r;
                    if (row < M) C[(size_t)row * N + col] = acc[i][j][r];
                }
            }
        }
    }
}

// ---------------------------------------------------------------------------
// Causal depthwise conv (K=4) + bias + SiLU.  Reads x half of xz.
// ---------------------------------------------------------------------------
__global__ void conv_silu_kernel(const float* __restrict__ xz,
                                 const float* __restrict__ conv_w,
                                 const float* __restrict__ conv_b,
                                 float* __restrict__ xc,
                                 unsigned short* __restrict__ xcb) {
    int idx = blockIdx.x * 256 + threadIdx.x;
    if (idx >= BL * EDIM) return;
    int e = idx % EDIM;
    int t = (idx / EDIM) % LDIM;
    int b = idx / (EDIM * LDIM);
    float s = conv_b[e];
    #pragma unroll
    for (int k = 0; k < KCONV; k++) {
        int tt = t - (KCONV - 1) + k;
        if (tt >= 0)
            s += conv_w[e * KCONV + k] *
                 xz[((size_t)(b * LDIM + tt)) * (2 * EDIM) + e];
    }
    float v = silu(s);
    xc[idx]  = v;
    xcb[idx] = f2bf(v);
}

// ---------------------------------------------------------------------------
// Extract dlow = dBC[:, :R] as bf16 for the dt-proj GEMM
// ---------------------------------------------------------------------------
__global__ void extract_dlow(const float* __restrict__ dBC,
                             unsigned short* __restrict__ dlow) {
    int i = blockIdx.x * 256 + threadIdx.x;
    if (i < BL * RRANK) {
        int m = i / RRANK, r = i % RRANK;
        dlow[i] = f2bf(dBC[(size_t)m * DBCW + r]);
    }
}

// ---------------------------------------------------------------------------
// delta = softplus(delta_raw + b_dt[e])  (in place)
// ---------------------------------------------------------------------------
__global__ void softplus_kernel(float* __restrict__ delta,
                                const float* __restrict__ b_dt) {
    int i = blockIdx.x * 256 + threadIdx.x;
    if (i < BL * EDIM) {
        int e = i % EDIM;
        float v = delta[i] + b_dt[e];
        delta[i] = (v > 20.0f) ? v : log1pf(__expf(v));
    }
}

// ---------------------------------------------------------------------------
// Selective scan: one lane per (b,e) channel, N=16 states in registers.
// B/C vectors (shared per timestep) staged into LDS in 64-step chunks.
// Grid: (EDIM/256, BDIM), block 256.
// ---------------------------------------------------------------------------
__launch_bounds__(256)
__global__ void scan_kernel(const float* __restrict__ delta,
                            const float* __restrict__ xc,
                            const float* __restrict__ dBC,
                            const float* __restrict__ A_log,
                            float* __restrict__ ys) {
    const int e = blockIdx.x * 256 + threadIdx.x;
    const int b = blockIdx.y;
    __shared__ float Bs[64][NSTATE];
    __shared__ float Cs[64][NSTATE];

    float Aneg[NSTATE];
    #pragma unroll
    for (int n = 0; n < NSTATE; n++)
        Aneg[n] = -__expf(A_log[(size_t)e * NSTATE + n]);

    float h[NSTATE];
    #pragma unroll
    for (int n = 0; n < NSTATE; n++) h[n] = 0.0f;

    for (int tc = 0; tc < LDIM; tc += 64) {
        // cooperative staging: 64 timesteps x 32 floats (B then C)
        for (int i = threadIdx.x; i < 64 * 2 * NSTATE; i += 256) {
            int tt = i >> 5;
            int c  = i & 31;
            float v = dBC[((size_t)(b * LDIM + tc + tt)) * DBCW + RRANK + c];
            if (c < NSTATE) Bs[tt][c] = v;
            else            Cs[tt][c - NSTATE] = v;
        }
        __syncthreads();
        for (int t = 0; t < 64; t++) {
            size_t idx = ((size_t)(b * LDIM + tc + t)) * EDIM + e;
            float d  = delta[idx];
            float dx = d * xc[idx];
            float y  = 0.0f;
            #pragma unroll
            for (int n = 0; n < NSTATE; n++) {
                float dA = __expf(d * Aneg[n]);
                h[n] = dA * h[n] + dx * Bs[t][n];
                y += h[n] * Cs[t][n];
            }
            ys[idx] = y;
        }
        __syncthreads();
    }
}

// ---------------------------------------------------------------------------
// y = (ys + D[e]*x) * silu(z)  -> bf16 for out-proj
// ---------------------------------------------------------------------------
__global__ void gate_kernel(const float* __restrict__ ys,
                            const float* __restrict__ xc,
                            const float* __restrict__ xz,
                            const float* __restrict__ Dp,
                            unsigned short* __restrict__ yb) {
    int i = blockIdx.x * 256 + threadIdx.x;
    if (i >= BL * EDIM) return;
    int e   = i % EDIM;
    int row = i / EDIM;
    float z = xz[(size_t)row * (2 * EDIM) + EDIM + e];
    float v = (ys[i] + Dp[e] * xc[i]) * silu(z);
    yb[i] = f2bf(v);
}

// ---------------------------------------------------------------------------
extern "C" void kernel_launch(void* const* d_in, const int* in_sizes, int n_in,
                              void* d_out, int out_size, void* d_ws, size_t ws_size,
                              hipStream_t stream) {
    const float* input  = (const float*)d_in[0];
    const float* W_in   = (const float*)d_in[1];
    const float* conv_w = (const float*)d_in[2];
    const float* conv_b = (const float*)d_in[3];
    const float* W_x    = (const float*)d_in[4];
    const float* W_dt   = (const float*)d_in[5];
    const float* b_dt   = (const float*)d_in[6];
    const float* A_log  = (const float*)d_in[7];
    const float* D_par  = (const float*)d_in[8];
    const float* W_out  = (const float*)d_in[9];
    float* out = (float*)d_out;

    char* p = (char*)d_ws;
    auto alloc = [&](size_t bytes) -> char* {
        char* q = p;
        p += (bytes + 255) & ~(size_t)255;
        return q;
    };
    unsigned short* in_b   = (unsigned short*)alloc((size_t)BL * DMODEL * 2);
    unsigned short* Win_b  = (unsigned short*)alloc((size_t)2 * EDIM * DMODEL * 2);
    unsigned short* Wx_b   = (unsigned short*)alloc((size_t)DBCW * EDIM * 2);
    unsigned short* Wdt_b  = (unsigned short*)alloc((size_t)EDIM * RRANK * 2);
    unsigned short* Wout_b = (unsigned short*)alloc((size_t)DMODEL * EDIM * 2);
    float*          xz     = (float*)alloc((size_t)BL * 2 * EDIM * 4);
    float*          xc     = (float*)alloc((size_t)BL * EDIM * 4);
    unsigned short* xc_b   = (unsigned short*)alloc((size_t)BL * EDIM * 2);
    float*          dBC    = (float*)alloc((size_t)BL * DBCW * 4);
    unsigned short* dlow_b = (unsigned short*)alloc((size_t)BL * RRANK * 2);
    float*          delta  = (float*)alloc((size_t)BL * EDIM * 4);
    float*          ysb    = (float*)alloc((size_t)BL * EDIM * 4);
    unsigned short* y_b    = (unsigned short*)alloc((size_t)BL * EDIM * 2);

    auto cvt = [&](const float* s, unsigned short* d, int n) {
        int n4 = n / 4;
        cvt_f32_to_bf16x4<<<(n4 + 255) / 256, 256, 0, stream>>>(s, d, n4);
    };

    // 1) operand conversions to bf16
    cvt(input, in_b,   BL * DMODEL);
    cvt(W_in,  Win_b,  2 * EDIM * DMODEL);
    cvt(W_x,   Wx_b,   DBCW * EDIM);
    cvt(W_dt,  Wdt_b,  EDIM * RRANK);
    cvt(W_out, Wout_b, DMODEL * EDIM);

    // 2) in-proj: xz = input @ W_in^T   (4096 x 4096, K=1024)
    gemm_bf16_wmma<<<dim3((2 * EDIM) / GBN, BL / GBM), 256, 0, stream>>>(
        in_b, Win_b, xz, BL, 2 * EDIM, DMODEL);

    // 3) causal depthwise conv + SiLU
    conv_silu_kernel<<<(BL * EDIM) / 256, 256, 0, stream>>>(
        xz, conv_w, conv_b, xc, xc_b);

    // 4) x-proj: dBC = xc @ W_x^T   (4096 x 96, K=2048)
    gemm_bf16_wmma<<<dim3((DBCW + GBN - 1) / GBN, BL / GBM), 256, 0, stream>>>(
        xc_b, Wx_b, dBC, BL, DBCW, EDIM);

    // 5) dt-proj: delta_raw = dlow @ W_dt^T  (4096 x 2048, K=64)
    extract_dlow<<<(BL * RRANK) / 256, 256, 0, stream>>>(dBC, dlow_b);
    gemm_bf16_wmma<<<dim3(EDIM / GBN, BL / GBM), 256, 0, stream>>>(
        dlow_b, Wdt_b, delta, BL, EDIM, RRANK);
    softplus_kernel<<<(BL * EDIM) / 256, 256, 0, stream>>>(delta, b_dt);

    // 6) selective scan
    scan_kernel<<<dim3(EDIM / 256, BDIM), 256, 0, stream>>>(
        delta, xc, dBC, A_log, ysb);

    // 7) gating
    gate_kernel<<<(BL * EDIM) / 256, 256, 0, stream>>>(ysb, xc, xz, D_par, y_b);

    // 8) out-proj: out = y @ W_out^T  (4096 x 1024, K=2048)
    gemm_bf16_wmma<<<dim3(DMODEL / GBN, BL / GBM), 256, 0, stream>>>(
        y_b, Wout_b, out, BL, DMODEL, EDIM);
}
